// SparseMOELayer_27702539059630
// MI455X (gfx1250) — matrix operand order repaired and checked by reading, not compiled, not run on hardware
//
#include <hip/hip_runtime.h>
#include <hip/hip_bf16.h>
#include <cmath>

#define N_TOK 65536
#define DIM   128
#define NEXP  16
#define TOPKK 3

typedef __attribute__((ext_vector_type(16))) __bf16 v16bf;
typedef __attribute__((ext_vector_type(8)))  float  v8f;

union FragU { uint4 q[2]; v16bf v; };

static __device__ __forceinline__ unsigned short f32_to_bf16(float f) {
  unsigned u = __builtin_bit_cast(unsigned, f);
  u += 0x7fffu + ((u >> 16) & 1u);           // round-to-nearest-even
  return (unsigned short)(u >> 16);
}
static __device__ __forceinline__ unsigned pack_bf16x2(float lo, float hi) {
  return (unsigned)f32_to_bf16(lo) | ((unsigned)f32_to_bf16(hi) << 16);
}

// ---------------------------------------------------------------------------
// Prep: We[e][d][f] f32 -> weT[e][f][d] bf16 ; be[e][f] f32 -> beT[f][e(0..31,pad0)] bf16
// ---------------------------------------------------------------------------
__global__ __launch_bounds__(256) void moe_prep(const float* __restrict__ We,
                                                const float* __restrict__ be,
                                                unsigned short* __restrict__ weT,
                                                unsigned short* __restrict__ beT) {
  int g = blockIdx.x * 256 + threadIdx.x;
  if (g < NEXP * DIM * DIM) {
    int f = g & 127, d = (g >> 7) & 127, e = g >> 14;  // coalesced read of We
    weT[(e << 14) + (f << 7) + d] = f32_to_bf16(We[g]);
  }
  if (g < DIM * 32) {
    int e = g & 31, f = g >> 5;
    float v = (e < NEXP) ? be[e * DIM + f] : 0.f;
    beT[f * 32 + e] = f32_to_bf16(v);
  }
}

// ---------------------------------------------------------------------------
// Gating: gate = x@Wg + bg + noise*softplus(x@Wn + bn); top-3 softmax -> comb
// Deterministic per-block column sums of comb -> partials (for importance loss)
// ---------------------------------------------------------------------------
__global__ __launch_bounds__(256) void moe_gate(const float* __restrict__ x,
                                                const float* __restrict__ noise,
                                                const float* __restrict__ Wg,
                                                const float* __restrict__ bg,
                                                const float* __restrict__ Wn,
                                                const float* __restrict__ bn,
                                                float* __restrict__ comb_out,
                                                float* __restrict__ partials) {
  __shared__ float sWg[DIM * NEXP];
  __shared__ float sWn[DIM * NEXP];
  __shared__ float sbg[NEXP], sbn[NEXP];
  __shared__ float sRed[256][NEXP];

  int t = threadIdx.x;
  for (int i = t; i < DIM * NEXP; i += 256) { sWg[i] = Wg[i]; sWn[i] = Wn[i]; }
  if (t < NEXP) { sbg[t] = bg[t]; sbn[t] = bn[t]; }
  __syncthreads();

  int n = blockIdx.x * 256 + t;
  const float4* xr = (const float4*)(x + (size_t)n * DIM);
  float g[NEXP], h[NEXP];
  #pragma unroll
  for (int e = 0; e < NEXP; ++e) { g[e] = 0.f; h[e] = 0.f; }

  for (int d0 = 0; d0 < DIM; d0 += 4) {
    float4 xv = xr[d0 >> 2];
    float xc[4] = {xv.x, xv.y, xv.z, xv.w};
    #pragma unroll
    for (int dd = 0; dd < 4; ++dd) {
      float c = xc[dd];
      const float* wg = &sWg[(d0 + dd) * NEXP];
      const float* wn = &sWn[(d0 + dd) * NEXP];
      #pragma unroll
      for (int e = 0; e < NEXP; ++e) {
        g[e] = fmaf(c, wg[e], g[e]);
        h[e] = fmaf(c, wn[e], h[e]);
      }
    }
  }

  const float4* nz4 = (const float4*)(noise + (size_t)n * NEXP);
  float4 na = nz4[0], nb = nz4[1], nc = nz4[2], nd = nz4[3];
  float nz[NEXP] = {na.x, na.y, na.z, na.w, nb.x, nb.y, nb.z, nb.w,
                    nc.x, nc.y, nc.z, nc.w, nd.x, nd.y, nd.z, nd.w};

  float v[NEXP];
  #pragma unroll
  for (int e = 0; e < NEXP; ++e) {
    float si = h[e] + sbn[e];
    float sp = fmaxf(si, 0.f) + log1pf(expf(-fabsf(si)));   // stable softplus
    v[e] = g[e] + sbg[e] + nz[e] * sp;
  }

  // top-3 (strict >, ties pick lowest index — matches lax.top_k)
  float vv[NEXP];
  #pragma unroll
  for (int e = 0; e < NEXP; ++e) vv[e] = v[e];
  int   ki[TOPKK];
  float kv[TOPKK];
  #pragma unroll
  for (int k = 0; k < TOPKK; ++k) {
    float b = -INFINITY; int bi = 0;
    #pragma unroll
    for (int e = 0; e < NEXP; ++e) { if (vv[e] > b) { b = vv[e]; bi = e; } }
    kv[k] = b; ki[k] = bi;
    #pragma unroll
    for (int e = 0; e < NEXP; ++e) { if (e == bi) vv[e] = -INFINITY; }
  }

  float m  = kv[0];
  float e0 = expf(kv[0] - m), e1 = expf(kv[1] - m), e2 = expf(kv[2] - m);
  float inv = 1.f / (e0 + e1 + e2);
  float w0 = e0 * inv, w1 = e1 * inv, w2 = e2 * inv;

  float cr[NEXP];
  #pragma unroll
  for (int e = 0; e < NEXP; ++e)
    cr[e] = (e == ki[0]) ? w0 : (e == ki[1]) ? w1 : (e == ki[2]) ? w2 : 0.f;

  float4* co = (float4*)(comb_out + (size_t)n * NEXP);
  co[0] = make_float4(cr[0],  cr[1],  cr[2],  cr[3]);
  co[1] = make_float4(cr[4],  cr[5],  cr[6],  cr[7]);
  co[2] = make_float4(cr[8],  cr[9],  cr[10], cr[11]);
  co[3] = make_float4(cr[12], cr[13], cr[14], cr[15]);

  #pragma unroll
  for (int e = 0; e < NEXP; ++e) sRed[t][e] = cr[e];
  __syncthreads();
  if (t < NEXP) {
    float s = 0.f;
    for (int r = 0; r < 256; ++r) s += sRed[r][t];   // fixed order: deterministic
    partials[blockIdx.x * NEXP + t] = s;
  }
}

// ---------------------------------------------------------------------------
// Importance loss: imp = column sums; loss = var(ddof=1)/mean^2 (deterministic)
// ---------------------------------------------------------------------------
__global__ void moe_loss(const float* __restrict__ partials, float* __restrict__ loss_out) {
  __shared__ float imp[NEXP];
  int t = threadIdx.x;
  if (t < NEXP) {
    float s = 0.f;
    for (int b = 0; b < N_TOK / 256; ++b) s += partials[b * NEXP + t];
    imp[t] = s;
  }
  __syncthreads();
  if (t == 0) {
    float mean = 0.f;
    for (int e = 0; e < NEXP; ++e) mean += imp[e];
    mean *= (1.f / NEXP);
    float var = 0.f;
    for (int e = 0; e < NEXP; ++e) { float d = imp[e] - mean; var += d * d; }
    var *= (1.f / (NEXP - 1));
    loss_out[0] = var / (mean * mean);
  }
}

// ---------------------------------------------------------------------------
// Expert compute: logits[n] = sum_e comb[n,e]*(x[n]@We[e]) + comb[n,:]@be
// 64-token tile/block, 8 waves; bf16 WMMA with f32 accum; fp32 combine scale.
// Double-buffered We staging via GLOBAL_LOAD_ASYNC_TO_LDS_B128 (ASYNCcnt).
// ---------------------------------------------------------------------------
struct MoeSmem {
  unsigned short xb[64][136];       // x tile, bf16, padded stride   (17408 B)
  unsigned short wt[2][128][136];   // We[e]^T tiles, double-buffered (69632 B)
  float          comb[64][16];      // combine weights                 (4096 B)
  unsigned short bet[128][32];      // be^T [f][e-padded]              (8192 B)
};                                  // total 99328 B (3 blocks/WGP in 320KB)

__global__ __launch_bounds__(256, 1) void moe_expert(const float* __restrict__ x,
                                                     const float* __restrict__ comb,
                                                     const unsigned short* __restrict__ weT,
                                                     const unsigned short* __restrict__ beT,
                                                     float* __restrict__ out) {
  __shared__ MoeSmem sm;
  int t    = threadIdx.x;
  int tok0 = blockIdx.x * 64;

  // thread's fixed slice of each We tile: row r, 64-bf16 half hh (128 B)
  int wr = t >> 1, wh = t & 1;

  // ---- issue async stage of expert 0 into buffer 0 (overlaps x conversion) --
  {
    unsigned long long g = (unsigned long long)(uintptr_t)(weT + (wr << 7) + wh * 64);
    unsigned l = (unsigned)(uintptr_t)&sm.wt[0][wr][wh * 64];
    #pragma unroll
    for (int i = 0; i < 8; ++i)   // offset applies to BOTH lds and global addr
      asm volatile("global_load_async_to_lds_b128 %0, %1, off offset:%2"
                   :: "v"(l), "v"(g), "n"(i * 16) : "memory");
  }

  // ---- stage x tile (f32 -> bf16) ----
  {
    int r = t >> 2, qh = t & 3;                                  // 64 rows x 4 quarters
    const float4* xr = (const float4*)(x + (size_t)(tok0 + r) * DIM + qh * 32);
    uint4* dst = (uint4*)&sm.xb[r][qh * 32];
    #pragma unroll
    for (int i = 0; i < 4; ++i) {
      float4 a = xr[2 * i], b = xr[2 * i + 1];
      uint4 q;
      q.x = pack_bf16x2(a.x, a.y);
      q.y = pack_bf16x2(a.z, a.w);
      q.z = pack_bf16x2(b.x, b.y);
      q.w = pack_bf16x2(b.z, b.w);
      dst[i] = q;
    }
  }
  // ---- stage comb tile ----
  {
    int r = t >> 2, s4 = t & 3;
    *(float4*)&sm.comb[r][s4 * 4] =
        *(const float4*)(comb + (size_t)(tok0 + r) * NEXP + s4 * 4);
  }
  // ---- stage be^T ----
  if (t < 128) {
    const uint4* src = (const uint4*)(beT + t * 32);
    uint4* dst = (uint4*)&sm.bet[t][0];
    #pragma unroll
    for (int i = 0; i < 4; ++i) dst[i] = src[i];
  }
  __syncthreads();

  int wid = t >> 5, lane = t & 31;
  int mt = wid & 3, nh = wid >> 2;      // wave -> (M tile, N half)
  int lm = lane & 15, lh = lane >> 4;

  // ---- bias term via WMMA: comb(16x16, K padded to 32) @ be(16x128) ----
  v8f acc[4];
  const v8f vzero = {0.f, 0.f, 0.f, 0.f, 0.f, 0.f, 0.f, 0.f};
  {
    FragU ca;
    const float* crow = &sm.comb[mt * 16 + lm][0];
    unsigned pk[4];
    #pragma unroll
    for (int p = 0; p < 4; ++p) {                 // A layout: K = lh*8 + 2p (+1)
      int e = lh * 8 + p * 2;
      pk[p] = pack_bf16x2(crow[e], crow[e + 1]);
    }
    ca.q[0] = make_uint4(pk[0], pk[1], pk[2], pk[3]);
    ca.q[1] = make_uint4(0u, 0u, 0u, 0u);         // K=16..31 zero pad
    #pragma unroll
    for (int j = 0; j < 4; ++j) {
      int f = (nh * 4 + j) * 16 + lm;
      FragU bb;                                   // B layout: K = lh*8 + 0..7 (+16)
      const uint4* bp = (const uint4*)&sm.bet[f][lh * 8];
      bb.q[0] = bp[0]; bb.q[1] = bp[2];
      acc[j] = __builtin_amdgcn_wmma_f32_16x16x32_bf16(
          false, ca.v, false, bb.v, (short)0, vzero, false, false);
    }
  }

  // ---- expert loop: wait(buf e) -> barrier -> async issue(buf e+1) -> WMMA(e)
  for (int e = 0; e < NEXP; ++e) {
    asm volatile("s_wait_asynccnt 0" ::: "memory"); // this wave's async done
    __syncthreads();                                // all waves' tiles visible;
                                                    // prior readers of buf(e+1) done
    if (e + 1 < NEXP) {
      unsigned long long g =
          (unsigned long long)(uintptr_t)(weT + ((e + 1) << 14) + (wr << 7) + wh * 64);
      unsigned l = (unsigned)(uintptr_t)&sm.wt[(e + 1) & 1][wr][wh * 64];
      #pragma unroll
      for (int i = 0; i < 8; ++i)
        asm volatile("global_load_async_to_lds_b128 %0, %1, off offset:%2"
                     :: "v"(l), "v"(g), "n"(i * 16) : "memory");
    }
    if (e + 2 < NEXP)                               // warm L2 two experts ahead
      __builtin_prefetch(weT + ((e + 2) << 14) + (t << 6), 0, 0);

    const unsigned short(*wt)[136] = sm.wt[e & 1];

    float cs[8];
    #pragma unroll
    for (int r = 0; r < 8; ++r)                    // C row M = r + 8*lh
      cs[r] = sm.comb[mt * 16 + r + 8 * lh][e];

    v8f y[4];
    #pragma unroll
    for (int j = 0; j < 4; ++j) y[j] = vzero;

    #pragma unroll
    for (int kt = 0; kt < 4; ++kt) {
      FragU a;                                     // A: K = kt*32 + lh*8 (+16)
      const uint4* ap = (const uint4*)&sm.xb[mt * 16 + lm][kt * 32 + lh * 8];
      a.q[0] = ap[0];
      a.q[1] = ap[2];                              // +32 bytes -> K+16 run
      #pragma unroll
      for (int j = 0; j < 4; ++j) {
        int f = (nh * 4 + j) * 16 + lm;
        FragU b;                                   // B: K = kt*32 + lh*8 (+16)
        const uint4* bp = (const uint4*)&wt[f][kt * 32 + lh * 8];
        b.q[0] = bp[0]; b.q[1] = bp[2];
        y[j] = __builtin_amdgcn_wmma_f32_16x16x32_bf16(
            false, a.v, false, b.v, (short)0, y[j], false, false);
      }
    }
    #pragma unroll
    for (int j = 0; j < 4; ++j) {
      #pragma unroll
      for (int r = 0; r < 8; ++r)
        acc[j][r] = fmaf(cs[r], y[j][r], acc[j][r]);  // fp32 combine scale
    }
  }

  // ---- write logits ----
  #pragma unroll
  for (int j = 0; j < 4; ++j) {
    int f = (nh * 4 + j) * 16 + lm;
    #pragma unroll
    for (int r = 0; r < 8; ++r) {
      int tokr = tok0 + mt * 16 + r + 8 * lh;
      out[(size_t)tokr * DIM + f] = acc[j][r];
    }
  }
}

// ---------------------------------------------------------------------------
extern "C" void kernel_launch(void* const* d_in, const int* in_sizes, int n_in,
                              void* d_out, int out_size, void* d_ws, size_t ws_size,
                              hipStream_t stream) {
  (void)in_sizes; (void)n_in; (void)out_size; (void)ws_size;
  const float* x     = (const float*)d_in[0];
  const float* noise = (const float*)d_in[1];
  const float* Wg    = (const float*)d_in[2];
  const float* bg    = (const float*)d_in[3];
  const float* Wn    = (const float*)d_in[4];
  const float* bn    = (const float*)d_in[5];
  const float* We    = (const float*)d_in[6];
  const float* be    = (const float*)d_in[7];

  float* logits = (float*)d_out;                        // [N*D]
  float* loss   = logits + (size_t)N_TOK * DIM;         // [1]
  float* combO  = loss + 1;                             // [N*E]

  char* ws = (char*)d_ws;
  unsigned short* weT = (unsigned short*)ws;            // 524288 B
  unsigned short* beT = (unsigned short*)(ws + 524288); // 8192 B
  float* partials     = (float*)(ws + 524288 + 8192);   // 16384 B

  moe_prep  <<<1024,        256, 0, stream>>>(We, be, weT, beT);
  moe_gate  <<<N_TOK / 256, 256, 0, stream>>>(x, noise, Wg, bg, Wn, bn, combO, partials);
  moe_loss  <<<1,           32,  0, stream>>>(partials, loss);
  moe_expert<<<N_TOK / 64,  256, 0, stream>>>(x, combO, weT, beT, logits);
}
